// EdgeNetDeeper_8177617731797
// MI455X (gfx1250) — matrix operand order repaired and checked by reading, not compiled, not run on hardware
//
#include <hip/hip_runtime.h>

typedef __attribute__((ext_vector_type(16))) _Float16 v16h;
typedef __attribute__((ext_vector_type(8)))  float    v8f;

#define BN_EPS 1e-5f

union AFrag { v16h v; unsigned int u[8]; };
union H2    { unsigned int u; _Float16 h[2]; };

// ---------------- elementwise / reduction helpers ----------------

__global__ void zero_f32_kernel(float* p, long n) {
  long i = (long)blockIdx.x * blockDim.x + threadIdx.x;
  if (i < n) p[i] = 0.f;
}

__global__ void count_dst_kernel(const int* __restrict__ dst, float* __restrict__ cnt, int E) {
  int e = blockIdx.x * blockDim.x + threadIdx.x;
  if (e < E) atomicAdd(&cnt[dst[e]], 1.0f);
}

// column stats (C=4) for the input BatchNorm over N rows
__global__ void stats_cols4_kernel(const float* __restrict__ x, float* sSum, float* sSq, int N) {
  __shared__ float ls[4], lq[4];
  if (threadIdx.x < 4) { ls[threadIdx.x] = 0.f; lq[threadIdx.x] = 0.f; }
  __syncthreads();
  long i = (long)blockIdx.x * blockDim.x + threadIdx.x;
  if (i < (long)N * 4) {
    float v = x[i];
    atomicAdd(&ls[i & 3], v);
    atomicAdd(&lq[i & 3], v * v);
  }
  __syncthreads();
  if (threadIdx.x < 4) {
    atomicAdd(&sSum[threadIdx.x], ls[threadIdx.x]);
    atomicAdd(&sSq[threadIdx.x], lq[threadIdx.x]);
  }
}

// mu = S/n, var = Q/n - mu^2 (biased), s = g*rsqrt(var+eps), t = bt - mu*s
__global__ void bn_finalize_kernel(const float* sSum, const float* sSq,
                                   const float* g, const float* bt,
                                   float* s, float* t, int C, float invCount) {
  int c = threadIdx.x;
  if (c < C) {
    float mu  = sSum[c] * invCount;
    float var = sSq[c] * invCount - mu * mu;
    float sc  = g[c] * rsqrtf(var + BN_EPS);
    s[c] = sc;
    t[c] = bt[c] - mu * sc;
  }
}

__global__ void node_affine_kernel(const float* __restrict__ x, const float* s, const float* t,
                                   float* __restrict__ h, long total, int C) {
  long i = (long)blockIdx.x * blockDim.x + threadIdx.x;
  if (i >= total) return;
  int c = (int)(i % C);
  h[i] = s[c] * x[i] + t[c];
}

// build edge features: row e = [ x_dst , x_src - x_dst , 0-pad ]  (f16, width CINPAD)
template<int C, int CINPAD>
__global__ void gather_kernel(const float* __restrict__ h, const int* __restrict__ src,
                              const int* __restrict__ dst, _Float16* __restrict__ X, int E) {
  long i = (long)blockIdx.x * blockDim.x + threadIdx.x;
  long tot = (long)E * CINPAD;
  if (i >= tot) return;
  long e = i / CINPAD;
  int  c = (int)(i % CINPAD);
  float v = 0.f;
  if (c < C) {
    v = h[(long)dst[e] * C + c];
  } else if (c < 2 * C) {
    int cc = c - C;
    v = h[(long)src[e] * C + cc] - h[(long)dst[e] * C + cc];
  }
  X[i] = (_Float16)v;
}

// repack fp32 weights [Cin][Cout] into f16 WMMA B fragments, zero-padded.
// B (32x16 f16): lane = n + 16*((k>>3)&1), slot = (k&7) + 8*(k>>4) within each 32-K chunk.
__global__ void prep_weights_kernel(const float* __restrict__ W, const float* __restrict__ b,
                                    _Float16* __restrict__ Wf, float* __restrict__ bpad,
                                    int Cin, int Cout, int CINPAD, int COUTPAD) {
  int KT = CINPAD >> 5;
  int i = blockIdx.x * blockDim.x + threadIdx.x;
  int tot = CINPAD * COUTPAD;
  if (i < tot) {
    int k = i / COUTPAD, n = i % COUTPAD;
    float v = (k < Cin && n < Cout) ? W[k * Cout + n] : 0.f;
    int kt = k >> 5, kk = k & 31, nt = n >> 4, nn = n & 15;
    int lane = nn + 16 * ((kk >> 3) & 1);
    int slot = (kk & 7) + 8 * (kk >> 4);
    Wf[(size_t)((nt * KT + kt) * 32 + lane) * 16 + slot] = (_Float16)v;
  }
  if (i < COUTPAD) bpad[i] = (i < Cout) ? b[i] : 0.f;
}

// ---------------- WMMA edge-batch GEMM ----------------
// X: [E][CIN] f16 row-major, Wf: fragment-packed, Y: [E][COUT] f16 row-major.
// Each wave handles 16 edges. Optional fused input-BN affine (+relu) applied in
// the A-fragment loader; optional output ReLU; optional per-channel sum/sumsq
// atomics for BatchNorm statistics. Output staged via per-wave LDS slab so the
// final global stores are coalesced b128.
template<int CIN, int COUT, bool RELU, bool STATS, bool IN_AFF, bool IN_RELU>
__global__ __launch_bounds__(256) void gemm_edge_kernel(
    const _Float16* __restrict__ X, const _Float16* __restrict__ Wf,
    const float* __restrict__ bias, _Float16* __restrict__ Y,
    const float* __restrict__ sA, const float* __restrict__ tA,
    float* __restrict__ sSum, float* __restrict__ sSq, int E) {
  constexpr int KT = CIN / 32, NT = COUT / 16;
  __shared__ __align__(16) _Float16 lsbuf[8 * 16 * COUT];
  int wave = threadIdx.x >> 5;
  int lane = threadIdx.x & 31;
  long tile = (long)blockIdx.x * 8 + wave;
  if (tile * 16 >= E) return;             // wave-uniform: EXEC stays all-ones for WMMA
  long base = tile * 16;
  _Float16* lw = lsbuf + wave * 16 * COUT;
  int m  = lane & 15;
  int hk = (lane >> 4) * 8;               // lanes 16-31 take K+8 / K+24 halves

  // A fragments (16x32 f16): dword pairs K{0,1},{2,3},{4,5},{6,7},{16,17}..{22,23}
  AFrag a[KT];
  const _Float16* xrow = X + (base + m) * CIN;
  #pragma unroll
  for (int kt = 0; kt < KT; ++kt) {
    #pragma unroll
    for (int d = 0; d < 8; ++d) {
      int halfoff = hk + (d < 4 ? 2 * d : 16 + 2 * (d - 4));
      H2 w;
      w.u = *(const unsigned int*)(xrow + kt * 32 + halfoff);
      if (IN_AFF) {                        // fused BatchNorm affine on the input
        int k0 = kt * 32 + halfoff;
        float v0 = (float)w.h[0] * sA[k0]     + tA[k0];
        float v1 = (float)w.h[1] * sA[k0 + 1] + tA[k0 + 1];
        if (IN_RELU) { v0 = fmaxf(v0, 0.f); v1 = fmaxf(v1, 0.f); }
        w.h[0] = (_Float16)v0;
        w.h[1] = (_Float16)v1;
      }
      a[kt].u[d] = w.u;
    }
  }

  int n = lane & 15;
  int rowhi = 8 * (lane >> 4);            // C/D: lane l row block M = r + 8*(l>>4)
  #pragma unroll
  for (int nt = 0; nt < NT; ++nt) {
    float b0 = bias[nt * 16 + n];
    v8f acc;
    #pragma unroll
    for (int r = 0; r < 8; ++r) acc[r] = b0;
    #pragma unroll
    for (int kt = 0; kt < KT; ++kt) {
      v16h bf = *(const v16h*)(Wf + (size_t)((nt * KT + kt) * 32 + lane) * 16);
      acc = __builtin_amdgcn_wmma_f32_16x16x32_f16(false, a[kt].v, false, bf,
                                                   (short)0, acc, false, false);
    }
    float sum = 0.f, sq = 0.f;
    #pragma unroll
    for (int r = 0; r < 8; ++r) {
      float v = acc[r];
      if (RELU) v = fmaxf(v, 0.f);
      lw[(r + rowhi) * COUT + nt * 16 + n] = (_Float16)v;
      if (STATS) { sum += v; sq += v * v; }
    }
    if (STATS) {
      sum += __shfl_xor(sum, 16, 32);     // combine lane pairs sharing channel n
      sq  += __shfl_xor(sq, 16, 32);
      if (lane < 16) {
        atomicAdd(&sSum[nt * 16 + n], sum);
        atomicAdd(&sSq[nt * 16 + n], sq);
      }
    }
  }

  // coalesced write-back: wave's 16xCOUT f16 slab is contiguous in Y
  constexpr int CH = COUT / 2;            // halves per lane
  _Float16* gout = Y + base * COUT;
  #pragma unroll
  for (int v = 0; v < COUT / 16; ++v) {
    uint4 d = *(const uint4*)(lw + lane * CH + v * 8);
    *(uint4*)(gout + lane * CH + v * 8) = d;
  }
}

template<int C, int CPAD>
__global__ void scatter_kernel(const _Float16* __restrict__ Ye, const int* __restrict__ dst,
                               float* __restrict__ acc, int E) {
  long i = (long)blockIdx.x * blockDim.x + threadIdx.x;
  long tot = (long)E * C;
  if (i >= tot) return;
  long e = i / C;
  int  c = (int)(i % C);
  atomicAdd(&acc[(long)dst[e] * C + c], (float)Ye[e * CPAD + c]);
}

// scatter-mean divide; final-BN affine commutes with the mean so it's applied here
template<int C, bool AFFINE>
__global__ void divide_affine_kernel(float* __restrict__ acc, const float* __restrict__ cnt,
                                     const float* s, const float* t,
                                     float* __restrict__ out, int N) {
  long i = (long)blockIdx.x * blockDim.x + threadIdx.x;
  if (i >= (long)N * C) return;
  long nidx = i / C;
  int  c = (int)(i % C);
  float d = cnt[nidx];
  d = d > 1.f ? d : 1.f;
  float v = acc[i] / d;
  if (AFFINE) v = s[c] * v + t[c];
  out[i] = v;
}

// ---------------- orchestration ----------------

extern "C" void kernel_launch(void* const* d_in, const int* in_sizes, int n_in,
                              void* d_out, int out_size, void* d_ws, size_t ws_size,
                              hipStream_t stream) {
  (void)n_in; (void)out_size; (void)ws_size;
  const float* x  = (const float*)d_in[0];
  const int*   ei = (const int*)d_in[1];
  const int N = in_sizes[0] / 4;
  const int E = in_sizes[1] / 2;
  const int* src = ei;
  const int* dst = ei + E;

  const float* P[48];
  for (int i = 2; i < 48; ++i) P[i] = (const float*)d_in[i];

  // workspace carve (256B aligned)
  char* ws = (char*)d_ws;
  size_t off = 0;
  auto carve = [&](size_t bytes) -> char* {
    char* p = ws + off;
    off = (off + bytes + 255) & ~(size_t)255;
    return p;
  };
  float*    stats = (float*)carve(256 * 4);          // sum[0:128] | sumsq[128:256]
  float*    sSum  = stats;
  float*    sSq   = stats + 128;
  float*    sB    = (float*)carve(128 * 4);
  float*    tB    = (float*)carve(128 * 4);
  float*    bpad  = (float*)carve(64 * 4);
  _Float16* wfrag = (_Float16*)carve((size_t)128 * 64 * 2);
  float*    cnt   = (float*)carve((size_t)N * 4);
  float*    nodeA = (float*)carve((size_t)N * 64 * 4);
  float*    nodeB = (float*)carve((size_t)N * 64 * 4);
  _Float16* edgeA = (_Float16*)carve((size_t)E * 128 * 2);
  _Float16* edgeB = (_Float16*)carve((size_t)E * 64 * 2);

  const int B = 256;
  auto g1 = [&](long n) { return dim3((unsigned)((n + B - 1) / B)); };
  dim3 ggemm((unsigned)((E / 16 + 7) / 8));
  float invE = 1.f / (float)E;

  // ---- input BatchNorm over nodes ----
  zero_f32_kernel<<<g1(256), B, 0, stream>>>(stats, 256);
  stats_cols4_kernel<<<g1((long)N * 4), B, 0, stream>>>(x, sSum, sSq, N);
  bn_finalize_kernel<<<1, 128, 0, stream>>>(sSum, sSq, P[2], P[3], sB, tB, 4, 1.f / (float)N);
  node_affine_kernel<<<g1((long)N * 4), B, 0, stream>>>(x, sB, tB, nodeA, (long)N * 4, 4);

  // ---- in-degree counts (shared by all 4 convs) ----
  zero_f32_kernel<<<g1(N), B, 0, stream>>>(cnt, N);
  count_dst_kernel<<<g1(E), B, 0, stream>>>(dst, cnt, E);

  // ===== conv1 (enc1): node 4 -> 32 =====
  gather_kernel<4, 32><<<g1((long)E * 32), B, 0, stream>>>(nodeA, src, dst, edgeA, E);
  prep_weights_kernel<<<g1(32 * 64), B, 0, stream>>>(P[4], P[5], wfrag, bpad, 8, 64, 32, 64);
  zero_f32_kernel<<<g1(256), B, 0, stream>>>(stats, 256);
  gemm_edge_kernel<32, 64, true, true, false, false><<<ggemm, 256, 0, stream>>>(
      edgeA, wfrag, bpad, edgeB, nullptr, nullptr, sSum, sSq, E);
  bn_finalize_kernel<<<1, 128, 0, stream>>>(sSum, sSq, P[6], P[7], sB, tB, 64, invE);
  prep_weights_kernel<<<g1(64 * 64), B, 0, stream>>>(P[8], P[9], wfrag, bpad, 64, 64, 64, 64);
  zero_f32_kernel<<<g1(256), B, 0, stream>>>(stats, 256);
  gemm_edge_kernel<64, 64, true, true, true, false><<<ggemm, 256, 0, stream>>>(
      edgeB, wfrag, bpad, edgeA, sB, tB, sSum, sSq, E);
  bn_finalize_kernel<<<1, 128, 0, stream>>>(sSum, sSq, P[10], P[11], sB, tB, 64, invE);
  prep_weights_kernel<<<g1(64 * 32), B, 0, stream>>>(P[12], P[13], wfrag, bpad, 64, 32, 64, 32);
  zero_f32_kernel<<<g1(256), B, 0, stream>>>(stats, 256);
  gemm_edge_kernel<64, 32, true, true, true, false><<<ggemm, 256, 0, stream>>>(
      edgeA, wfrag, bpad, edgeB, sB, tB, sSum, sSq, E);
  bn_finalize_kernel<<<1, 128, 0, stream>>>(sSum, sSq, P[14], P[15], sB, tB, 32, invE);
  zero_f32_kernel<<<g1((long)N * 32), B, 0, stream>>>(nodeB, (long)N * 32);
  scatter_kernel<32, 32><<<g1((long)E * 32), B, 0, stream>>>(edgeB, dst, nodeB, E);
  divide_affine_kernel<32, true><<<g1((long)N * 32), B, 0, stream>>>(nodeB, cnt, sB, tB, nodeB, N);

  // ===== conv2 (enc2): node 32 -> 2 =====
  gather_kernel<32, 64><<<g1((long)E * 64), B, 0, stream>>>(nodeB, src, dst, edgeA, E);
  prep_weights_kernel<<<g1(64 * 32), B, 0, stream>>>(P[16], P[17], wfrag, bpad, 64, 32, 64, 32);
  zero_f32_kernel<<<g1(256), B, 0, stream>>>(stats, 256);
  gemm_edge_kernel<64, 32, true, true, false, false><<<ggemm, 256, 0, stream>>>(
      edgeA, wfrag, bpad, edgeB, nullptr, nullptr, sSum, sSq, E);
  bn_finalize_kernel<<<1, 128, 0, stream>>>(sSum, sSq, P[18], P[19], sB, tB, 32, invE);
  prep_weights_kernel<<<g1(32 * 32), B, 0, stream>>>(P[20], P[21], wfrag, bpad, 32, 32, 32, 32);
  zero_f32_kernel<<<g1(256), B, 0, stream>>>(stats, 256);
  gemm_edge_kernel<32, 32, true, true, true, false><<<ggemm, 256, 0, stream>>>(
      edgeB, wfrag, bpad, edgeA, sB, tB, sSum, sSq, E);
  bn_finalize_kernel<<<1, 128, 0, stream>>>(sSum, sSq, P[22], P[23], sB, tB, 32, invE);
  prep_weights_kernel<<<g1(32 * 16), B, 0, stream>>>(P[24], P[25], wfrag, bpad, 32, 2, 32, 16);
  zero_f32_kernel<<<g1(256), B, 0, stream>>>(stats, 256);
  gemm_edge_kernel<32, 16, true, true, true, false><<<ggemm, 256, 0, stream>>>(
      edgeA, wfrag, bpad, edgeB, sB, tB, sSum, sSq, E);
  bn_finalize_kernel<<<1, 128, 0, stream>>>(sSum, sSq, P[26], P[27], sB, tB, 2, invE);
  zero_f32_kernel<<<g1((long)N * 2), B, 0, stream>>>(nodeA, (long)N * 2);
  scatter_kernel<2, 16><<<g1((long)E * 2), B, 0, stream>>>(edgeB, dst, nodeA, E);
  divide_affine_kernel<2, true><<<g1((long)N * 2), B, 0, stream>>>(nodeA, cnt, sB, tB, nodeA, N);

  // ===== conv3 (dec1): node 2 -> 64  (Lin ReLU | Lin BN ReLU | Lin ReLU BN) =====
  gather_kernel<2, 32><<<g1((long)E * 32), B, 0, stream>>>(nodeA, src, dst, edgeA, E);
  prep_weights_kernel<<<g1(32 * 32), B, 0, stream>>>(P[28], P[29], wfrag, bpad, 4, 32, 32, 32);
  gemm_edge_kernel<32, 32, true, false, false, false><<<ggemm, 256, 0, stream>>>(
      edgeA, wfrag, bpad, edgeB, nullptr, nullptr, sSum, sSq, E);
  prep_weights_kernel<<<g1(32 * 32), B, 0, stream>>>(P[30], P[31], wfrag, bpad, 32, 32, 32, 32);
  zero_f32_kernel<<<g1(256), B, 0, stream>>>(stats, 256);
  gemm_edge_kernel<32, 32, false, true, false, false><<<ggemm, 256, 0, stream>>>(
      edgeB, wfrag, bpad, edgeA, nullptr, nullptr, sSum, sSq, E);
  bn_finalize_kernel<<<1, 128, 0, stream>>>(sSum, sSq, P[32], P[33], sB, tB, 32, invE);
  prep_weights_kernel<<<g1(32 * 64), B, 0, stream>>>(P[34], P[35], wfrag, bpad, 32, 64, 32, 64);
  zero_f32_kernel<<<g1(256), B, 0, stream>>>(stats, 256);
  // fused BN(affine)+ReLU on the input of L3 (dec1's Lin->BN->ReLU ordering)
  gemm_edge_kernel<32, 64, true, true, true, true><<<ggemm, 256, 0, stream>>>(
      edgeA, wfrag, bpad, edgeB, sB, tB, sSum, sSq, E);
  bn_finalize_kernel<<<1, 128, 0, stream>>>(sSum, sSq, P[36], P[37], sB, tB, 64, invE);
  zero_f32_kernel<<<g1((long)N * 64), B, 0, stream>>>(nodeB, (long)N * 64);
  scatter_kernel<64, 64><<<g1((long)E * 64), B, 0, stream>>>(edgeB, dst, nodeB, E);
  divide_affine_kernel<64, true><<<g1((long)N * 64), B, 0, stream>>>(nodeB, cnt, sB, tB, nodeB, N);

  // ===== conv4 (dec2): node 64 -> 4 (final; last Lin has no ReLU/BN) =====
  gather_kernel<64, 128><<<g1((long)E * 128), B, 0, stream>>>(nodeB, src, dst, edgeA, E);
  prep_weights_kernel<<<g1(128 * 64), B, 0, stream>>>(P[38], P[39], wfrag, bpad, 128, 64, 128, 64);
  zero_f32_kernel<<<g1(256), B, 0, stream>>>(stats, 256);
  gemm_edge_kernel<128, 64, true, true, false, false><<<ggemm, 256, 0, stream>>>(
      edgeA, wfrag, bpad, edgeB, nullptr, nullptr, sSum, sSq, E);
  bn_finalize_kernel<<<1, 128, 0, stream>>>(sSum, sSq, P[40], P[41], sB, tB, 64, invE);
  prep_weights_kernel<<<g1(64 * 64), B, 0, stream>>>(P[42], P[43], wfrag, bpad, 64, 64, 64, 64);
  zero_f32_kernel<<<g1(256), B, 0, stream>>>(stats, 256);
  gemm_edge_kernel<64, 64, true, true, true, false><<<ggemm, 256, 0, stream>>>(
      edgeB, wfrag, bpad, edgeA, sB, tB, sSum, sSq, E);
  bn_finalize_kernel<<<1, 128, 0, stream>>>(sSum, sSq, P[44], P[45], sB, tB, 64, invE);
  prep_weights_kernel<<<g1(64 * 16), B, 0, stream>>>(P[46], P[47], wfrag, bpad, 64, 4, 64, 16);
  gemm_edge_kernel<64, 16, false, false, true, false><<<ggemm, 256, 0, stream>>>(
      edgeA, wfrag, bpad, edgeB, sB, tB, sSum, sSq, E);
  float* out = (float*)d_out;
  zero_f32_kernel<<<g1((long)N * 4), B, 0, stream>>>(out, (long)N * 4);
  scatter_kernel<4, 16><<<g1((long)E * 4), B, 0, stream>>>(edgeB, dst, out, E);
  divide_affine_kernel<4, false><<<g1((long)N * 4), B, 0, stream>>>(out, cnt, nullptr, nullptr, out, N);
}
